// NVFP4InferenceLinear_38242388803949
// MI455X (gfx1250) — matrix operand order repaired and checked by reading, not compiled, not run on hardware
//
#include <hip/hip_runtime.h>

typedef __attribute__((ext_vector_type(16))) __bf16        v16bf;
typedef __attribute__((ext_vector_type(2)))  __bf16        v2bf;
typedef __attribute__((ext_vector_type(8)))  float         v8f;
typedef __attribute__((ext_vector_type(2)))  float         v2f;
typedef __attribute__((ext_vector_type(4)))  unsigned int  v4u;
typedef __attribute__((ext_vector_type(4)))  float         v4f;
typedef __attribute__((ext_vector_type(4)))  int           v4i;

#define K_DIM   4096
#define N_DIM   4096
#define M_DIM   8192
#define KPACK   (K_DIM / 2)      // int32 elements per weight row (1 byte each)
#define NBLK    (K_DIM / 16)     // fp4 scale blocks per weight row
#define TM      128
#define TN      128
#define TK      64
#define LDW     72               // LDS row pitch in bf16 elems (144B = 9*16B, conflict-skewed)

union FragU { struct { v4u lo, hi; } p; v16bf v; };

// e2m1 decode: sign|e2|m1 -> {0,.5,1,1.5,2,3,4,6} with sign
__device__ __forceinline__ float e2m1_decode(int c) {
    int e = (c >> 1) & 3;
    int m = c & 1;
    float mag = (e == 0) ? 0.5f * (float)m
                         : (float)(2 + m) * 0.25f * (float)(1 << e);
    return (c & 8) ? -mag : mag;
}

// Pack two f32 -> packed bf16 pair (targets v_cvt_pk_bf16_f32)
__device__ __forceinline__ unsigned int pack_bf16(float a, float b) {
    v2f f; f.x = a; f.y = b;
    v2bf h = __builtin_convertvector(f, v2bf);
    return __builtin_bit_cast(unsigned int, h);
}

__global__ __launch_bounds__(256)
void nvfp4_linear_wmma(const float* __restrict__ x,
                       const int*   __restrict__ w,      // packed fp4, 1 byte per int32
                       const float* __restrict__ wsf,    // [N, NBLK]
                       const float* __restrict__ gs,     // [1]
                       const float* __restrict__ bias,   // [N]
                       float*       __restrict__ out)
{
    __shared__ __align__(16) unsigned short As[TM][LDW]; // x tile, bf16 bits, [m][k]
    __shared__ __align__(16) unsigned short Bs[TN][LDW]; // dequant W tile, bf16 bits, [n][k]

    const int t     = threadIdx.x;
    const int lane  = t & 31;
    const int waveM = (t >> 5) >> 2;   // 0..1
    const int waveN = (t >> 5) & 3;    // 0..3
    const int rowBase = blockIdx.y * TM;
    const int colBase = blockIdx.x * TN;

    const float gscale = gs[0];

    // x staging map: 8 iters of (row = aRow+16i, 4 floats at aCol)
    const int aRow = t >> 4;           // 0..15
    const int aCol = (t & 15) * 4;     // 0..60
    // w staging map: 4 iters of (row = bRow+32i, 4 int32 at bCol) -> 8 bf16
    const int bRow = t >> 3;           // 0..31
    const int bCol = (t & 7) * 4;      // int32 index in chunk, 0..28

    v4f   aReg[8];
    v4i   bReg[4];
    float sReg[4];

    v8f acc[4][2];
    const v8f zero8 = {0.f, 0.f, 0.f, 0.f, 0.f, 0.f, 0.f, 0.f};
#pragma unroll
    for (int mi = 0; mi < 4; ++mi)
#pragma unroll
        for (int ni = 0; ni < 2; ++ni) acc[mi][ni] = zero8;

    auto loadA = [&](int kBase) {
#pragma unroll
        for (int i = 0; i < 8; ++i) {
            const size_t r = (size_t)(rowBase + aRow + 16 * i);
            aReg[i] = *(const v4f*)(x + r * K_DIM + kBase + aCol);
        }
    };
    auto loadB = [&](int kBase) {
        const int kp = kBase >> 1;                    // packed-int index base
        const int blk = (kBase >> 4) + (bCol >> 3);   // scale block for this int4
#pragma unroll
        for (int i = 0; i < 4; ++i) {
            const size_t r = (size_t)(colBase + bRow + 32 * i);
            bReg[i] = *(const v4i*)(w + r * KPACK + kp + bCol);
            sReg[i] = wsf[r * NBLK + blk] * gscale;
        }
    };
    auto stageA = [&]() {
#pragma unroll
        for (int i = 0; i < 8; ++i) {
            uint2 p;
            p.x = pack_bf16(aReg[i].x, aReg[i].y);
            p.y = pack_bf16(aReg[i].z, aReg[i].w);
            *(uint2*)(&As[aRow + 16 * i][aCol]) = p;
        }
    };
    auto stageB = [&]() {
#pragma unroll
        for (int i = 0; i < 4; ++i) {
            v4u pk;
#pragma unroll
            for (int j = 0; j < 4; ++j) {
                const int byte = bReg[i][j] & 0xFF;
                const float lo = e2m1_decode(byte & 0xF) * sReg[i];
                const float hi = e2m1_decode((byte >> 4) & 0xF) * sReg[i];
                pk[j] = pack_bf16(lo, hi);
            }
            *(v4u*)(&Bs[bRow + 32 * i][bCol * 2]) = pk;
        }
    };

    // A fragment (16x32 bf16): lanes 0-15 K=[kk,kk+8)+[kk+16,kk+24); lanes 16-31 shifted by 8
    auto loadFragA = [&](int row, int kk) -> v16bf {
        FragU f;
        const int r = row + (lane & 15);
        const int c = kk + ((lane >> 4) << 3);
        f.p.lo = *(const v4u*)(&As[r][c]);
        f.p.hi = *(const v4u*)(&As[r][c + 16]);
        return f.v;
    };
    // B fragment (32x16 bf16): lane holds col N=lane&15; lanes 0-15 K=[kk,kk+8)+[kk+16,kk+24); lanes 16-31 shifted by 8
    auto loadFragB = [&](int row, int kk) -> v16bf {
        FragU f;
        const int r = row + (lane & 15);
        const int c = kk + ((lane >> 4) << 3);
        f.p.lo = *(const v4u*)(&Bs[r][c]);
        f.p.hi = *(const v4u*)(&Bs[r][c + 16]);
        return f.v;
    };

    loadA(0);
    loadB(0);

    for (int kBase = 0; kBase < K_DIM; kBase += TK) {
        stageA();
        stageB();
        __syncthreads();

        if (kBase + TK < K_DIM) {
            loadA(kBase + TK);
            loadB(kBase + TK);
        }
        if (kBase + 2 * TK < K_DIM) {
            __builtin_prefetch(x + (size_t)(rowBase + aRow) * K_DIM + kBase + 2 * TK + aCol, 0, 0);
            __builtin_prefetch(w + (size_t)(colBase + bRow) * KPACK + ((kBase + 2 * TK) >> 1) + bCol, 0, 0);
        }

#pragma unroll
        for (int ks = 0; ks < 2; ++ks) {
            const int kk = ks * 32;
            v16bf aF[4], bF[2];
#pragma unroll
            for (int mi = 0; mi < 4; ++mi) aF[mi] = loadFragA(waveM * 64 + mi * 16, kk);
#pragma unroll
            for (int ni = 0; ni < 2; ++ni) bF[ni] = loadFragB(waveN * 32 + ni * 16, kk);
#pragma unroll
            for (int mi = 0; mi < 4; ++mi)
#pragma unroll
                for (int ni = 0; ni < 2; ++ni)
                    acc[mi][ni] = __builtin_amdgcn_wmma_f32_16x16x32_bf16(
                        false, aF[mi], false, bF[ni], (short)0, acc[mi][ni], false, false);
        }
        __syncthreads();
    }

    // Epilogue: C 16x16 f32 layout -> VGPR r: lanes 0-15 M=r, lanes 16-31 M=8+r; N=lane&15
#pragma unroll
    for (int ni = 0; ni < 2; ++ni) {
        const int col = colBase + waveN * 32 + ni * 16 + (lane & 15);
        const float bv = bias[col];
#pragma unroll
        for (int mi = 0; mi < 4; ++mi) {
            const int row0 = rowBase + waveM * 64 + mi * 16 + ((lane >> 4) << 3);
#pragma unroll
            for (int r = 0; r < 8; ++r) {
                out[(size_t)(row0 + r) * N_DIM + col] = acc[mi][ni][r] + bv;
            }
        }
    }
}

extern "C" void kernel_launch(void* const* d_in, const int* in_sizes, int n_in,
                              void* d_out, int out_size, void* d_ws, size_t ws_size,
                              hipStream_t stream) {
    const float* x    = (const float*)d_in[0];
    const int*   w    = (const int*)d_in[1];
    const float* wsf  = (const float*)d_in[2];
    const float* gs   = (const float*)d_in[3];
    const float* bias = (const float*)d_in[4];
    float* out = (float*)d_out;

    dim3 grid(N_DIM / TN, M_DIM / TM);   // (32, 64)
    nvfp4_linear_wmma<<<grid, 256, 0, stream>>>(x, w, wsf, gs, bias, out);
}
